// PerfectAudioLayer_48808008351949
// MI455X (gfx1250) — matrix operand order repaired and checked by reading, not compiled, not run on hardware
//
#include <hip/hip_runtime.h>
#include <math.h>

// ---------------- types ----------------
typedef _Float16 h8  __attribute__((ext_vector_type(8)));
typedef _Float16 h16 __attribute__((ext_vector_type(16)));
typedef float    f8  __attribute__((ext_vector_type(8)));

union FragH { h16 v; h8 h[2]; };

// ---------------- problem constants ----------------
#define NWIN      65536
#define WIN       256
#define MIN_FREQS 202        // int(256*(1-0.3*0.7))
#define MIN_TIMES 51773      // int(65536*(1-0.3*0.7))
#define ETH       0.05f
#define TWO_PI_OVER_256 0.024543692606170259758f

// ---------------- GEMM tiling ----------------
#define BM 32                // rows per strip
#define BN 64                // cols per col-group
#define ASTR 264             // LDS row stride (halves), mult of 8 (16B), 132 dwords -> bank spread
#define BSTR 264
#define NSTRIPS (NWIN / BM)  // 2048

// ============ K_init: zero scratch accumulators (every call, deterministic) ============
__global__ void k_init(float* Eg, int* hist, unsigned* absmax, unsigned* tmax,
                       int* tcount, float* cutoff, int* mode, float* g0) {
  int t = threadIdx.x;
  Eg[t] = 0.f;
  for (int i = t; i < 4096; i += 256) hist[i] = 0;
  if (t == 0) { *absmax = 0u; *tmax = 0u; *tcount = 0; *cutoff = 0.f; *mode = 0; *g0 = 0.f; }
}

// ============ K0: forward DFT matrix, transposed: Bt[f][k], f<256 cos, f>=256 sin ============
__global__ void k_dft(_Float16* __restrict__ Bt) {
  int f = blockIdx.x;          // 0..511
  int k = threadIdx.x;         // 0..255
  int prod = (k * (f & 255)) & 255;
  float ang = (float)prod * TWO_PI_OVER_256;
  float v = (f < 256) ? cosf(ang) : sinf(ang);
  Bt[f * 256 + k] = (_Float16)v;
}

// ============ K1: global abs-max + per-window sum-of-squares (Parseval time energy) ============
__global__ void __launch_bounds__(256) k_absmax_tenergy(const float* __restrict__ W,
                                                        float* __restrict__ tE,
                                                        unsigned* __restrict__ absmax) {
  __shared__ float smx[8];
  int tid = threadIdx.x, w = tid >> 5, lane = tid & 31;
  int wid = blockIdx.x * 8 + w;                  // one wave per window
  const float* base = W + wid * WIN;
  float sq = 0.f, mx = 0.f;
#pragma unroll
  for (int j = 0; j < 8; ++j) {
    float v = base[lane + 32 * j];
    sq += v * v;
    mx = fmaxf(mx, fabsf(v));
  }
#pragma unroll
  for (int off = 16; off > 0; off >>= 1) {
    sq += __shfl_xor(sq, off, 32);
    mx = fmaxf(mx, __shfl_xor(mx, off, 32));
  }
  if (lane == 0) { tE[wid] = sq; smx[w] = mx; }
  __syncthreads();
  if (tid == 0) {
    float m = smx[0];
#pragma unroll
    for (int i = 1; i < 8; ++i) m = fmaxf(m, smx[i]);
    atomicMax(absmax, __float_as_uint(m));
  }
}

// ============ K2: spectrum power via WMMA: E[f] += (W@C)^2 + (W@S)^2 ============
__global__ void __launch_bounds__(256) k_specenergy(const float* __restrict__ W,
                                                    const _Float16* __restrict__ Bt,
                                                    float* __restrict__ Eg) {
  __shared__ _Float16 Ash[BM * ASTR];
  __shared__ _Float16 Bsh[BN * BSTR];
  __shared__ float Eblk[512];
  int tid = threadIdx.x, w = tid >> 5, lane = tid & 31;
  int ln = lane & 15, lh = lane >> 4;
  int mt = w >> 2, nt = w & 3;                  // 2 m-tiles x 4 n-tiles = 8 waves
  for (int i = tid; i < 512; i += 256) Eblk[i] = 0.f;

  for (int strip = blockIdx.x; strip < NSTRIPS; strip += gridDim.x) {
    __syncthreads();
    if (strip + gridDim.x < NSTRIPS)
      __builtin_prefetch(W + (size_t)(strip + gridDim.x) * BM * WIN, 0, 3);
    // stage A strip (32x256) f32->f16, float4-coalesced
    const float4* Wg4 = (const float4*)(W + strip * (BM * WIN));
    for (int u = tid; u < BM * WIN / 4; u += 256) {
      float4 v = Wg4[u];
      int row = u >> 6, c4 = (u & 63) * 4;
      _Float16* p = &Ash[row * ASTR + c4];
      p[0] = (_Float16)v.x; p[1] = (_Float16)v.y; p[2] = (_Float16)v.z; p[3] = (_Float16)v.w;
    }
    for (int cg = 0; cg < 8; ++cg) {            // 8 col-groups of 64 over the 512 (cos|sin) cols
      __syncthreads();
      const h8* src = (const h8*)(Bt + (cg * BN) * 256);
      for (int u = tid; u < BN * WIN / 8; u += 256) {
        int n = u >> 5, kc = u & 31;
        *(h8*)&Bsh[n * BSTR + kc * 8] = src[u];
      }
      __syncthreads();
      f8 acc = {0.f,0.f,0.f,0.f,0.f,0.f,0.f,0.f};
      const _Float16* arow = &Ash[(mt * 16 + ln) * ASTR];
      const _Float16* brow = &Bsh[(nt * 16 + ln) * BSTR];
#pragma unroll
      for (int ks = 0; ks < 8; ++ks) {          // K = 256 in 8 x 32
        int k0 = ks * 32;
        FragH a, b;
        a.h[0] = *(const h8*)(arow + k0 + lh * 8);        // K = k0+kb .. +7
        a.h[1] = *(const h8*)(arow + k0 + 16 + lh * 8);   // K = k0+16+kb .. +7
        b.h[0] = *(const h8*)(brow + k0 + lh * 8);        // K = k0+kb .. +7
        b.h[1] = *(const h8*)(brow + k0 + 16 + lh * 8);   // K = k0+16+kb .. +7
        acc = __builtin_amdgcn_wmma_f32_16x16x32_f16(false, a.v, false, b.v,
                                                     (short)0, acc, false, false);
      }
      float e = 0.f;
#pragma unroll
      for (int r = 0; r < 8; ++r) e += acc[r] * acc[r];   // sum over this lane's 8 rows
      atomicAdd(&Eblk[cg * BN + nt * 16 + ln], e);
    }
  }
  __syncthreads();
  for (int i = tid; i < 512; i += 256)
    atomicAdd(&Eg[i & 255], Eblk[i]);           // fold sin cols onto their frequency
}

// ============ K3: freq mask (threshold / exact top-202), g[d], residual matrix R (f16) ============
__global__ void k_freqmask(const float* __restrict__ Eg, _Float16* __restrict__ Rt,
                           float* __restrict__ g0p) {
  __shared__ float se[256], sred[256], fm[256], gsh[256];
  __shared__ int scnt;
  int t = threadIdx.x;
  float e = Eg[t];
  se[t] = e; sred[t] = e;
  __syncthreads();
  for (int off = 128; off > 0; off >>= 1) {
    if (t < off) sred[t] = fmaxf(sred[t], sred[t + off]);
    __syncthreads();
  }
  float th = ETH * sred[0];
  int pass = (e > th) ? 1 : 0;
  if (t == 0) scnt = 0;
  __syncthreads();
  atomicAdd(&scnt, pass);
  __syncthreads();
  int mask;
  if (scnt < MIN_FREQS) {                       // exact top-k, earliest-index tie-break
    int rank = 0;
    for (int j = 0; j < 256; ++j) {
      float ej = se[j];
      rank += (ej > e) || (ej == e && j < t);
    }
    mask = rank < MIN_FREQS;
  } else {
    mask = pass;
  }
  fm[t] = mask ? 1.f : 0.f;
  __syncthreads();
  float acc = 0.f;                              // g[d] = (1/256) sum_f fm[f] cos(2*pi*f*d/256)
  for (int f = 0; f < 256; ++f) {
    int p = (f * t) & 255;
    acc += fm[f] * cosf((float)p * TWO_PI_OVER_256);
  }
  gsh[t] = acc * (1.f / 256.f);
  __syncthreads();
  if (t == 0) *g0p = gsh[0];
  // residual R = M - g0*I (symmetric circulant), stored row-major [col][k]
  for (int k = 0; k < 256; ++k) {
    float v = (k == t) ? 0.f : gsh[(k - t) & 255];
    Rt[t * 256 + k] = (_Float16)v;
  }
}

// ============ K4a: max of time energies ============
__global__ void k_tmax(const float* __restrict__ tE, unsigned* __restrict__ tmax) {
  __shared__ float sred[256];
  int t = threadIdx.x;
  float e = tE[blockIdx.x * 256 + t];
  sred[t] = e;
  __syncthreads();
  for (int off = 128; off > 0; off >>= 1) {
    if (t < off) sred[t] = fmaxf(sred[t], sred[t + off]);
    __syncthreads();
  }
  if (t == 0) atomicMax(tmax, __float_as_uint(sred[0]));
}

// ============ K4b: count above threshold + 4096-bin histogram ============
__global__ void k_thist(const float* __restrict__ tE, const unsigned* __restrict__ tmax,
                        int* __restrict__ tcount, int* __restrict__ hist) {
  __shared__ int sh[4096];
  __shared__ int scnt;
  int t = threadIdx.x;
  for (int i = t; i < 4096; i += 256) sh[i] = 0;
  if (t == 0) scnt = 0;
  __syncthreads();
  float mx = __uint_as_float(*tmax);
  float th = ETH * mx;
  float inv = (mx > 0.f) ? 4096.f / mx : 0.f;
  float e = tE[blockIdx.x * 256 + t];
  if (e > th) atomicAdd(&scnt, 1);
  int bin = (int)(e * inv); if (bin > 4095) bin = 4095;
  atomicAdd(&sh[bin], 1);
  __syncthreads();
  if (t == 0) atomicAdd(tcount, scnt);
  for (int i = t; i < 4096; i += 256) if (sh[i]) atomicAdd(&hist[i], sh[i]);
}

// ============ K4c: decide cutoff (threshold vs histogram-select fallback) ============
__global__ void k_tcut(const int* __restrict__ tcount, const int* __restrict__ hist,
                       const unsigned* __restrict__ tmax, float* __restrict__ cutoff,
                       int* __restrict__ mode) {
  if (threadIdx.x == 0) {
    float mx = __uint_as_float(*tmax);
    if (*tcount >= MIN_TIMES) { *cutoff = ETH * mx; *mode = 0; }   // strict >
    else {
      int acc = 0, b = 4095;
      for (; b > 0; --b) { acc += hist[b]; if (acc >= MIN_TIMES) break; }
      *cutoff = (float)b * mx * (1.f / 4096.f);                    // bin lower edge
      *mode = 1;                                                   // inclusive >=
    }
  }
}

// ============ K4d: per-window scale s[t] = keep ? 1/norm : 0 ============
__global__ void k_scale(const float* __restrict__ tE, const float* __restrict__ cutoff,
                        const int* __restrict__ mode, const unsigned* __restrict__ absmax,
                        float* __restrict__ s) {
  int g = blockIdx.x * blockDim.x + threadIdx.x;
  float e = tE[g];
  float c = *cutoff;
  int keep = (*mode) ? (e >= c) : (e > c);
  float am = __uint_as_float(*absmax);
  float norm = (am > 1e-10f) ? am : 1.f;
  s[g] = keep ? (1.f / norm) : 0.f;
}

// ============ K5: reconstruction: out = s[t]*(g0*W + W_f16 @ R) via WMMA ============
__global__ void __launch_bounds__(256) k_recon(const float* __restrict__ W,
                                               const _Float16* __restrict__ Rt,
                                               const float* __restrict__ s,
                                               const float* __restrict__ g0p,
                                               float* __restrict__ out) {
  __shared__ _Float16 Ash[BM * ASTR];
  __shared__ _Float16 Bsh[BN * BSTR];
  int tid = threadIdx.x, w = tid >> 5, lane = tid & 31;
  int ln = lane & 15, lh = lane >> 4;
  int mt = w >> 2, nt = w & 3;
  float g0 = *g0p;

  for (int strip = blockIdx.x; strip < NSTRIPS; strip += gridDim.x) {
    __syncthreads();
    if (strip + gridDim.x < NSTRIPS)
      __builtin_prefetch(W + (size_t)(strip + gridDim.x) * BM * WIN, 0, 3);
    const float4* Wg4 = (const float4*)(W + strip * (BM * WIN));
    for (int u = tid; u < BM * WIN / 4; u += 256) {
      float4 v = Wg4[u];
      int row = u >> 6, c4 = (u & 63) * 4;
      _Float16* p = &Ash[row * ASTR + c4];
      p[0] = (_Float16)v.x; p[1] = (_Float16)v.y; p[2] = (_Float16)v.z; p[3] = (_Float16)v.w;
    }
    for (int cg = 0; cg < 4; ++cg) {            // 4 col-groups of 64 over 256 output cols
      __syncthreads();
      const h8* src = (const h8*)(Rt + (cg * BN) * 256);
      for (int u = tid; u < BN * WIN / 8; u += 256) {
        int n = u >> 5, kc = u & 31;
        *(h8*)&Bsh[n * BSTR + kc * 8] = src[u];
      }
      __syncthreads();
      f8 acc = {0.f,0.f,0.f,0.f,0.f,0.f,0.f,0.f};
      const _Float16* arow = &Ash[(mt * 16 + ln) * ASTR];
      const _Float16* brow = &Bsh[(nt * 16 + ln) * BSTR];
#pragma unroll
      for (int ks = 0; ks < 8; ++ks) {
        int k0 = ks * 32;
        FragH a, b;
        a.h[0] = *(const h8*)(arow + k0 + lh * 8);
        a.h[1] = *(const h8*)(arow + k0 + 16 + lh * 8);
        b.h[0] = *(const h8*)(brow + k0 + lh * 8);
        b.h[1] = *(const h8*)(brow + k0 + 16 + lh * 8);
        acc = __builtin_amdgcn_wmma_f32_16x16x32_f16(false, a.v, false, b.v,
                                                     (short)0, acc, false, false);
      }
      // epilogue: C layout -> row = r + 8*lh within tile, col = ln
      int colg = cg * BN + nt * 16 + ln;
      int rowbase = strip * BM + mt * 16 + lh * 8;
#pragma unroll
      for (int r = 0; r < 8; ++r) {
        int row = rowbase + r;
        float sv = s[row];
        float wv = W[row * WIN + colg];
        out[row * WIN + colg] = sv * (g0 * wv + acc[r]);
      }
    }
  }
}

// ================================ launcher ================================
extern "C" void kernel_launch(void* const* d_in, const int* in_sizes, int n_in,
                              void* d_out, int out_size, void* d_ws, size_t ws_size,
                              hipStream_t stream) {
  (void)in_sizes; (void)n_in; (void)out_size; (void)ws_size;
  const float* W = (const float*)d_in[0];
  float* out = (float*)d_out;
  char* ws = (char*)d_ws;

  _Float16* Bt     = (_Float16*)(ws);            // 512*256 f16 = 262144 B
  _Float16* Rt     = (_Float16*)(ws + 262144);   // 256*256 f16 = 131072 B
  float*    tE     = (float*)(ws + 393216);      // 65536 f32   = 262144 B
  float*    s      = (float*)(ws + 655360);      // 65536 f32   = 262144 B
  float*    Eg     = (float*)(ws + 917504);      // 256 f32
  int*      hist   = (int*)(ws + 918528);        // 4096 i32
  unsigned* absmax = (unsigned*)(ws + 934912);
  unsigned* tmax   = (unsigned*)(ws + 934916);
  int*      tcount = (int*)(ws + 934920);
  float*    cutoff = (float*)(ws + 934924);
  int*      mode   = (int*)(ws + 934928);
  float*    g0     = (float*)(ws + 934932);

  k_init<<<1, 256, 0, stream>>>(Eg, hist, absmax, tmax, tcount, cutoff, mode, g0);
  k_dft<<<512, 256, 0, stream>>>(Bt);
  k_absmax_tenergy<<<NWIN / 8, 256, 0, stream>>>(W, tE, absmax);
  k_specenergy<<<256, 256, 0, stream>>>(W, Bt, Eg);
  k_freqmask<<<1, 256, 0, stream>>>(Eg, Rt, g0);
  k_tmax<<<256, 256, 0, stream>>>(tE, tmax);
  k_thist<<<256, 256, 0, stream>>>(tE, tmax, tcount, hist);
  k_tcut<<<1, 32, 0, stream>>>(tcount, hist, tmax, cutoff, mode);
  k_scale<<<256, 256, 0, stream>>>(tE, cutoff, mode, absmax, s);
  k_recon<<<256, 256, 0, stream>>>(W, Rt, s, g0, out);
}
